// NodeMPNNLayer_19971597927006
// MI455X (gfx1250) — compile-verified, matrix operand order checked by reading, not run on hardware
//
#include <hip/hip_runtime.h>
#include <hip/hip_bf16.h>
#include <stddef.h>

// ---------------------------------------------------------------------------
// CDNA5 (gfx1250) fused MPNN layer: bf16 WMMA edge-MLP + atomic scatter-mean,
// then fused LN/FF/LN node update, also on bf16 WMMA.
// Round 2: batch B-fragment LDS loads per k-step so WMMAs issue back-to-back
// (one s_wait per 8 WMMAs instead of per WMMA), + global prefetch of the
// next edge_attr tile.
// ---------------------------------------------------------------------------

typedef __attribute__((ext_vector_type(16))) __bf16 v16bf;
typedef __attribute__((ext_vector_type(8)))  __bf16 v8bf;
typedef __attribute__((ext_vector_type(8)))  float  v8f;

#define N_NODES 10000
#define N_EDGES 640000
#define DIM     128
#define HIDDIM  256

#define WAVES   8
#define EDGE_TILES (N_EDGES / 16)   // 40000
#define NODE_TILES (N_NODES / 16)   // 625

// K-dim padding so LDS row strides are == 4 banks (mod 64): conflict-free
#define KP1 392   // 384 + 8  (392*2B = 196 DW, 196 % 64 == 4)
#define KP2 136   // 128 + 8  ( 68 DW % 64 == 4)
#define KPH 264   // 256 + 8  (132 DW % 64 == 4)

// ---- edge kernel LDS layout (bytes) ---------------------------------------
#define E_OFF_W1T  0
#define E_SZ_W1T   (128 * KP1 * 2)          // 100352
#define E_OFF_W2T  (E_OFF_W1T + E_SZ_W1T)
#define E_SZ_WSQ   (128 * KP2 * 2)          // 34816
#define E_OFF_W3T  (E_OFF_W2T + E_SZ_WSQ)
#define E_OFF_B1   (E_OFF_W3T + E_SZ_WSQ)
#define E_OFF_B2   (E_OFF_B1 + 128 * 4)
#define E_OFF_B3   (E_OFF_B2 + 128 * 4)
#define E_OFF_WAVE (E_OFF_B3 + 128 * 4)     // 171520
#define E_PER_WAVE (16 * KP1 * 2 + 16 * KP2 * 2 + 128)  // 17024
#define E_SMEM     (E_OFF_WAVE + WAVES * E_PER_WAVE)    // 307712 <= 320KB

// ---- node kernel LDS layout (bytes) ---------------------------------------
#define N_OFF_F1T  0
#define N_SZ_F1T   (256 * KP2 * 2)          // 69632
#define N_OFF_F2T  (N_OFF_F1T + N_SZ_F1T)
#define N_SZ_F2T   (128 * KPH * 2)          // 67584
#define N_OFF_BF1  (N_OFF_F2T + N_SZ_F2T)
#define N_OFF_BF2  (N_OFF_BF1 + 256 * 4)
#define N_OFF_G0   (N_OFF_BF2 + 128 * 4)
#define N_OFF_BE0  (N_OFF_G0 + 128 * 4)
#define N_OFF_G1   (N_OFF_BE0 + 128 * 4)
#define N_OFF_BE1  (N_OFF_G1 + 128 * 4)
#define N_OFF_WAVE (N_OFF_BE1 + 128 * 4)    // 140800
#define N_PER_WAVE (16 * KP2 * 2 + 16 * KPH * 2 + 16 * 128 * 4) // 20992
#define N_SMEM     (N_OFF_WAVE + WAVES * N_PER_WAVE)            // 308736

// wave-local LDS write->read ordering (CDNA5 split dep counters)
#define WAVE_SYNC() asm volatile("s_wait_dscnt 0x0" ::: "memory")

// Build a 16-element bf16 WMMA fragment per ISA 7.12.2 (K 0..7 | 16..23 halves)
static __device__ __forceinline__ v16bf load_frag(const __bf16* p) {
    v8bf lo = *(const v8bf*)p;          // K = base .. base+7
    v8bf hi = *(const v8bf*)(p + 16);   // K = base+16 .. base+23
    return __builtin_shufflevector(lo, hi, 0, 1, 2, 3, 4, 5, 6, 7,
                                           8, 9, 10, 11, 12, 13, 14, 15);
}

static __device__ __forceinline__ float xor1_sum(float v) {
    // ds_swizzle group-of-32: xor_mask=1, and_mask=0x1f -> pairwise exchange
    int o = __builtin_amdgcn_ds_swizzle(__float_as_int(v), 0x041f);
    return v + __int_as_float(o);
}

static __device__ __forceinline__ v8f wmma_bf16(v16bf a, v16bf b, v8f c) {
    return __builtin_amdgcn_wmma_f32_16x16x32_bf16(
        false, a, false, b, (short)0, c, false, false);
}

// ---------------------------------------------------------------------------
__global__ void zero_f32_kernel(float* __restrict__ p, int n) {
    int i = blockIdx.x * 256 + threadIdx.x;
    if (i < n) p[i] = 0.0f;
}

// ---------------------------------------------------------------------------
// Edge MLP: m = W3( relu( W2( relu( W1 [x_dst | e | x_src] + b1 )) + b2 )) + b3
// scatter-add into sums[dst], counts into cnt[dst].
// One wave == one 16-edge tile (persistent).
// ---------------------------------------------------------------------------
__global__ void __launch_bounds__(256)
edge_mlp_kernel(const float* __restrict__ x,
                const long long* __restrict__ eidx,   // [2, E] int64 (src,dst)
                const float* __restrict__ eattr,
                const float* __restrict__ W1, const float* __restrict__ b1,
                const float* __restrict__ W2, const float* __restrict__ b2,
                const float* __restrict__ W3, const float* __restrict__ b3,
                float* __restrict__ sums, float* __restrict__ cnt)
{
    extern __shared__ __align__(16) char smem[];
    __bf16* W1t = (__bf16*)(smem + E_OFF_W1T);
    __bf16* W2t = (__bf16*)(smem + E_OFF_W2T);
    __bf16* W3t = (__bf16*)(smem + E_OFF_W3T);
    float*  b1s = (float*)(smem + E_OFF_B1);
    float*  b2s = (float*)(smem + E_OFF_B2);
    float*  b3s = (float*)(smem + E_OFF_B3);

    const int tid = threadIdx.x;
    // stage weights (transposed: Wt[n][k], bf16) once per workgroup
    for (int i = tid; i < 384 * 128; i += 256) {
        int k = i >> 7, n = i & 127;
        W1t[n * KP1 + k] = (__bf16)W1[i];
    }
    for (int i = tid; i < 128 * 128; i += 256) {
        int k = i >> 7, n = i & 127;
        W2t[n * KP2 + k] = (__bf16)W2[i];
        W3t[n * KP2 + k] = (__bf16)W3[i];
    }
    for (int i = tid; i < 128; i += 256) { b1s[i] = b1[i]; b2s[i] = b2[i]; b3s[i] = b3[i]; }
    __syncthreads();

    const int wave = tid >> 5, lane = tid & 31;
    char* wb = smem + E_OFF_WAVE + wave * E_PER_WAVE;
    __bf16* inA = (__bf16*)wb;                          // 16 x KP1 bf16
    __bf16* act = (__bf16*)(wb + 16 * KP1 * 2);         // 16 x KP2 bf16
    int* srcI   = (int*)(wb + 16 * KP1 * 2 + 16 * KP2 * 2);
    int* dstI   = srcI + 16;

    const int m16 = lane & 15;
    const int kh8 = (lane >> 4) << 3;   // K half offset == C-row half offset

    const v8f vzero = {0.f, 0.f, 0.f, 0.f, 0.f, 0.f, 0.f, 0.f};

    for (int tile = blockIdx.x * WAVES + wave; tile < EDGE_TILES;
         tile += gridDim.x * WAVES) {
        const int e0 = tile * 16;

        // prefetch next tile's edge_attr stream (global_prefetch_b8):
        // 32 lanes x 256B apart cover the full 16x128 fp32 = 8KB tile
        {
            int tn = tile + gridDim.x * WAVES;
            if (tn < EDGE_TILES)
                __builtin_prefetch(eattr + (size_t)tn * 16 * DIM + lane * 64, 0, 3);
        }

        if (lane < 16) {
            srcI[lane] = (int)eidx[e0 + lane];
            dstI[lane] = (int)eidx[N_EDGES + e0 + lane];
        }
        WAVE_SYNC();

        // gather 16 x [x_dst | edge_attr | x_src] (fp32 -> bf16 LDS tile)
        for (int it = lane; it < 16 * 96; it += 32) {
            int r = it / 96;
            int c = (it % 96) * 4;
            const float* sp;
            if (c < 128)      sp = x + (size_t)dstI[r] * DIM + c;
            else if (c < 256) sp = eattr + (size_t)(e0 + r) * DIM + (c - 128);
            else              sp = x + (size_t)srcI[r] * DIM + (c - 256);
            float4 f = *(const float4*)sp;
            __bf16* dp = inA + r * KP1 + c;
            dp[0] = (__bf16)f.x; dp[1] = (__bf16)f.y;
            dp[2] = (__bf16)f.z; dp[3] = (__bf16)f.w;
        }
        WAVE_SYNC();

        v8f acc[8];
        v16bf bfr[8];

        // ---- layer 1: [16x384] @ [384x128] ------------------------------
        #pragma unroll
        for (int nt = 0; nt < 8; ++nt) acc[nt] = vzero;
        #pragma unroll
        for (int ks = 0; ks < 12; ++ks) {
            v16bf a = load_frag(inA + m16 * KP1 + ks * 32 + kh8);
            #pragma unroll
            for (int nt = 0; nt < 8; ++nt)          // batch all B loads
                bfr[nt] = load_frag(W1t + (nt * 16 + m16) * KP1 + ks * 32 + kh8);
            #pragma unroll
            for (int nt = 0; nt < 8; ++nt)          // back-to-back WMMAs
                acc[nt] = wmma_bf16(a, bfr[nt], acc[nt]);
        }
        #pragma unroll
        for (int nt = 0; nt < 8; ++nt) {
            int n = nt * 16 + m16;
            float bb = b1s[n];
            #pragma unroll
            for (int r = 0; r < 8; ++r) {
                float v = fmaxf(acc[nt][r] + bb, 0.f);
                act[(r + kh8) * KP2 + n] = (__bf16)v;
            }
        }
        WAVE_SYNC();

        // ---- layer 2: [16x128] @ [128x128] ------------------------------
        #pragma unroll
        for (int nt = 0; nt < 8; ++nt) acc[nt] = vzero;
        #pragma unroll
        for (int ks = 0; ks < 4; ++ks) {
            v16bf a = load_frag(act + m16 * KP2 + ks * 32 + kh8);
            #pragma unroll
            for (int nt = 0; nt < 8; ++nt)
                bfr[nt] = load_frag(W2t + (nt * 16 + m16) * KP2 + ks * 32 + kh8);
            #pragma unroll
            for (int nt = 0; nt < 8; ++nt)
                acc[nt] = wmma_bf16(a, bfr[nt], acc[nt]);
        }
        #pragma unroll
        for (int nt = 0; nt < 8; ++nt) {
            int n = nt * 16 + m16;
            float bb = b2s[n];
            #pragma unroll
            for (int r = 0; r < 8; ++r) {
                float v = fmaxf(acc[nt][r] + bb, 0.f);
                act[(r + kh8) * KP2 + n] = (__bf16)v;
            }
        }
        WAVE_SYNC();

        // ---- layer 3 + scatter-add -------------------------------------
        #pragma unroll
        for (int nt = 0; nt < 8; ++nt) acc[nt] = vzero;
        #pragma unroll
        for (int ks = 0; ks < 4; ++ks) {
            v16bf a = load_frag(act + m16 * KP2 + ks * 32 + kh8);
            #pragma unroll
            for (int nt = 0; nt < 8; ++nt)
                bfr[nt] = load_frag(W3t + (nt * 16 + m16) * KP2 + ks * 32 + kh8);
            #pragma unroll
            for (int nt = 0; nt < 8; ++nt)
                acc[nt] = wmma_bf16(a, bfr[nt], acc[nt]);
        }
        #pragma unroll
        for (int nt = 0; nt < 8; ++nt) {
            int n = nt * 16 + m16;
            float bb = b3s[n];
            #pragma unroll
            for (int r = 0; r < 8; ++r) {
                int m = r + kh8;
                atomicAdd(sums + (size_t)dstI[m] * DIM + n, acc[nt][r] + bb);
            }
        }
        if (lane < 16) atomicAdd(cnt + dstI[lane], 1.0f);
    }
}

// ---------------------------------------------------------------------------
// Node update: h = LN(x + sums/max(cnt,1)); out = LN(h + FF(h))
// One wave == one 16-node tile (persistent).
// ---------------------------------------------------------------------------
__global__ void __launch_bounds__(256)
node_update_kernel(const float* __restrict__ x,
                   const float* __restrict__ F1, const float* __restrict__ bf1,
                   const float* __restrict__ F2, const float* __restrict__ bf2,
                   const float* __restrict__ g0, const float* __restrict__ be0,
                   const float* __restrict__ g1, const float* __restrict__ be1,
                   const float* __restrict__ sums, const float* __restrict__ cnt,
                   float* __restrict__ out)
{
    extern __shared__ __align__(16) char smem[];
    __bf16* F1t = (__bf16*)(smem + N_OFF_F1T);   // [256][KP2]
    __bf16* F2t = (__bf16*)(smem + N_OFF_F2T);   // [128][KPH]
    float* bf1s = (float*)(smem + N_OFF_BF1);
    float* bf2s = (float*)(smem + N_OFF_BF2);
    float* g0s  = (float*)(smem + N_OFF_G0);
    float* be0s = (float*)(smem + N_OFF_BE0);
    float* g1s  = (float*)(smem + N_OFF_G1);
    float* be1s = (float*)(smem + N_OFF_BE1);

    const int tid = threadIdx.x;
    for (int i = tid; i < 128 * 256; i += 256) {     // F1 [128][256]
        int k = i >> 8, n = i & 255;
        F1t[n * KP2 + k] = (__bf16)F1[i];
    }
    for (int i = tid; i < 256 * 128; i += 256) {     // F2 [256][128]
        int k = i >> 7, n = i & 127;
        F2t[n * KPH + k] = (__bf16)F2[i];
    }
    for (int i = tid; i < 256; i += 256) bf1s[i] = bf1[i];
    if (tid < 128) {
        bf2s[tid] = bf2[tid];
        g0s[tid] = g0[tid]; be0s[tid] = be0[tid];
        g1s[tid] = g1[tid]; be1s[tid] = be1[tid];
    }
    __syncthreads();

    const int wave = tid >> 5, lane = tid & 31;
    char* wb = smem + N_OFF_WAVE + wave * N_PER_WAVE;
    __bf16* hA   = (__bf16*)wb;                                  // 16 x KP2 bf16
    __bf16* hidA = (__bf16*)(wb + 16 * KP2 * 2);                 // 16 x KPH bf16
    float*  hF   = (float*)(wb + 16 * KP2 * 2 + 16 * KPH * 2);   // 16 x 128 f32

    const int m16 = lane & 15;
    const int kh8 = (lane >> 4) << 3;
    const int row = lane >> 1, par = lane & 1;   // 2 lanes per LN row
    const v8f vzero = {0.f, 0.f, 0.f, 0.f, 0.f, 0.f, 0.f, 0.f};

    for (int tile = blockIdx.x * WAVES + wave; tile < NODE_TILES;
         tile += gridDim.x * WAVES) {
        const int v0 = tile * 16;

        // pre-norm residual input: x + sums/max(cnt,1)
        for (int it = lane; it < 16 * DIM; it += 32) {
            int r = it >> 7, c = it & 127;
            int v = v0 + r;
            float dh = sums[(size_t)v * DIM + c] / fmaxf(cnt[v], 1.f);
            hF[r * DIM + c] = x[(size_t)v * DIM + c] + dh;
        }
        WAVE_SYNC();

        // LayerNorm 0 -> h (kept f32 in hF for residual, bf16 in hA for WMMA)
        {
            float s = 0.f, sq = 0.f;
            #pragma unroll 8
            for (int i = 0; i < 64; ++i) {
                float v = hF[row * DIM + par + 2 * i];
                s += v; sq += v * v;
            }
            s = xor1_sum(s); sq = xor1_sum(sq);
            float mean = s * (1.f / DIM);
            float var  = sq * (1.f / DIM) - mean * mean;
            float rs   = rsqrtf(var + 1e-6f);
            #pragma unroll 8
            for (int i = 0; i < 64; ++i) {
                int c = par + 2 * i;
                float h = (hF[row * DIM + c] - mean) * rs * g0s[c] + be0s[c];
                hF[row * DIM + c] = h;
                hA[row * KP2 + c] = (__bf16)h;
            }
        }
        WAVE_SYNC();

        v8f acc[8];
        v16bf bfr[8];

        // FF layer 1: [16x128] @ [128x256], relu -> hidA  (two halves of 8 nt)
        #pragma unroll
        for (int half = 0; half < 2; ++half) {
            #pragma unroll
            for (int nt = 0; nt < 8; ++nt) acc[nt] = vzero;
            #pragma unroll
            for (int ks = 0; ks < 4; ++ks) {
                v16bf a = load_frag(hA + m16 * KP2 + ks * 32 + kh8);
                #pragma unroll
                for (int nt = 0; nt < 8; ++nt)
                    bfr[nt] = load_frag(F1t + ((half * 8 + nt) * 16 + m16) * KP2
                                        + ks * 32 + kh8);
                #pragma unroll
                for (int nt = 0; nt < 8; ++nt)
                    acc[nt] = wmma_bf16(a, bfr[nt], acc[nt]);
            }
            #pragma unroll
            for (int nt = 0; nt < 8; ++nt) {
                int n = (half * 8 + nt) * 16 + m16;
                float bb = bf1s[n];
                #pragma unroll
                for (int r = 0; r < 8; ++r) {
                    float v = fmaxf(acc[nt][r] + bb, 0.f);
                    hidA[(r + kh8) * KPH + n] = (__bf16)v;
                }
            }
        }
        WAVE_SYNC();

        // FF layer 2: [16x256] @ [256x128] + bias + residual -> hF
        #pragma unroll
        for (int nt = 0; nt < 8; ++nt) acc[nt] = vzero;
        #pragma unroll
        for (int ks = 0; ks < 8; ++ks) {
            v16bf a = load_frag(hidA + m16 * KPH + ks * 32 + kh8);
            #pragma unroll
            for (int nt = 0; nt < 8; ++nt)
                bfr[nt] = load_frag(F2t + (nt * 16 + m16) * KPH + ks * 32 + kh8);
            #pragma unroll
            for (int nt = 0; nt < 8; ++nt)
                acc[nt] = wmma_bf16(a, bfr[nt], acc[nt]);
        }
        #pragma unroll
        for (int nt = 0; nt < 8; ++nt) {
            int n = nt * 16 + m16;
            float bb = bf2s[n];
            #pragma unroll
            for (int r = 0; r < 8; ++r) {
                int m = r + kh8;
                hF[m * DIM + n] = acc[nt][r] + bb + hF[m * DIM + n];
            }
        }
        WAVE_SYNC();

        // LayerNorm 1 -> out
        {
            float s = 0.f, sq = 0.f;
            #pragma unroll 8
            for (int i = 0; i < 64; ++i) {
                float v = hF[row * DIM + par + 2 * i];
                s += v; sq += v * v;
            }
            s = xor1_sum(s); sq = xor1_sum(sq);
            float mean = s * (1.f / DIM);
            float var  = sq * (1.f / DIM) - mean * mean;
            float rs   = rsqrtf(var + 1e-6f);
            #pragma unroll 8
            for (int i = 0; i < 64; ++i) {
                int c = par + 2 * i;
                float o = (hF[row * DIM + c] - mean) * rs * g1s[c] + be1s[c];
                out[(size_t)(v0 + row) * DIM + c] = o;
            }
        }
    }
}

// ---------------------------------------------------------------------------
extern "C" void kernel_launch(void* const* d_in, const int* in_sizes, int n_in,
                              void* d_out, int out_size, void* d_ws, size_t ws_size,
                              hipStream_t stream) {
    const float*     x     = (const float*)d_in[0];
    const long long* eidx  = (const long long*)d_in[1];
    const float*     eattr = (const float*)d_in[2];
    const float*     W1 = (const float*)d_in[3];
    const float*     b1 = (const float*)d_in[4];
    const float*     W2 = (const float*)d_in[5];
    const float*     b2 = (const float*)d_in[6];
    const float*     W3 = (const float*)d_in[7];
    const float*     b3 = (const float*)d_in[8];
    const float*     F1 = (const float*)d_in[9];
    const float*     bf1 = (const float*)d_in[10];
    const float*     F2 = (const float*)d_in[11];
    const float*     bf2 = (const float*)d_in[12];
    const float*     g0 = (const float*)d_in[13];
    const float*     be0 = (const float*)d_in[14];
    const float*     g1 = (const float*)d_in[15];
    const float*     be1 = (const float*)d_in[16];
    float* out  = (float*)d_out;

    float* sums = (float*)d_ws;                       // [N_NODES][DIM]
    float* cnt  = sums + (size_t)N_NODES * DIM;       // [N_NODES]

    const int ztot = N_NODES * DIM + N_NODES;
    zero_f32_kernel<<<(ztot + 255) / 256, 256, 0, stream>>>(sums, ztot);

    edge_mlp_kernel<<<512, 256, E_SMEM, stream>>>(
        x, eidx, eattr, W1, b1, W2, b2, W3, b3, sums, cnt);

    node_update_kernel<<<80, 256, N_SMEM, stream>>>(
        x, F1, bf1, F2, bf2, g0, be0, g1, be1, sums, cnt, out);
}
